// dgcnn_segm_64037962383749
// MI455X (gfx1250) — matrix operand order repaired and checked by reading, not compiled, not run on hardware
//
#include <hip/hip_runtime.h>
#include <hip/hip_bf16.h>
#include <hip/hip_fp16.h>

// ---------------------------------------------------------------------------
// Types for CDNA5 WMMA
// ---------------------------------------------------------------------------
typedef _Float16 h16 __attribute__((ext_vector_type(16)));
typedef _Float16 h8v __attribute__((ext_vector_type(8)));
typedef float f8 __attribute__((ext_vector_type(8)));

#define BM 64
#define BN 64
#define BKK 32
#define LDSROW 40   // 32 halves payload + 8 pad; 80B row keeps 16B alignment

__device__ __forceinline__ void atomicMaxFloat(float* addr, float val) {
    unsigned int* ai = (unsigned int*)addr;
    unsigned int old = *ai;
    while (true) {
        float f = __uint_as_float(old);
        if (f >= val) break;
        unsigned int assumed = old;
        old = atomicCAS(ai, assumed, __float_as_uint(val));
        if (old == assumed) break;
    }
}

// ---------------------------------------------------------------------------
// WMMA GEMM:  Y(M,N) = A(M,K) * W(N,K)^T with fused epilogues.
// Contract: M % 64 == 0, Kdim % 32 == 0, K1 % 32 == 0, row strides % 8 == 0.
// HASK1: first K1 cols of A broadcast from Abc (B x K1)  (layer-8 concat).
// NFULL: N is a multiple of 64 -> no column guards anywhere (all but layer 11).
//  epi 0: y = lrelu(y*s[n]+b[n]) -> out16 (M x N f16)
//  epi 1: y = lrelu(y*s[n]+b[n]) -> atomic float max into outAux (B x N)
//  epi 2: y raw                  -> outAux as (B, N, rowsPerBatch) f32
// LDS tiles in fragment-ready permuted layout:
//  A: [row][hs][h] where k = (h&7) + 8*hs + 16*(h>>3)   (hs = lane>>4)
//  B: [col][hs][h] where k = (h&7) + 8*hs + 16*(h>>3)                  (hs = lane>>4)
// -> each fragment = two contiguous 16B LDS reads per lane.
// ---------------------------------------------------------------------------
template <bool HASK1, bool NFULL>
__global__ void __launch_bounds__(256)
gemm_wmma_kernel(const _Float16* __restrict__ A, const _Float16* __restrict__ Abc, int K1,
                 const _Float16* __restrict__ W, const float* __restrict__ scale,
                 const float* __restrict__ bias,
                 _Float16* __restrict__ out16, float* __restrict__ outAux,
                 int M, int N, int Kdim, int epi, int rowsPerBatch) {
    __shared__ __align__(16) _Float16 AsP[BM * LDSROW];
    __shared__ __align__(16) _Float16 BsP[BN * LDSROW];

    const int tid  = threadIdx.x;
    const int lane = tid & 31;
    const int wv   = tid >> 5;             // 8 waves
    const int blockM = blockIdx.x * BM;
    const int blockN = blockIdx.y * BN;
    const int Astride = Kdim - (HASK1 ? K1 : 0);
    const int mr  = (wv >> 1) * 16;        // wave row offset
    const int nb0 = (wv & 1) * 32;         // wave col offset (2 subtiles)
    const int bb  = blockM / rowsPerBatch; // uniform across tile (rpb % 64 == 0)

    // staging geometry (loop-invariant)
    const int ar   = tid >> 2;             // 0..63 row
    const int ac0  = (tid & 3) * 8;        // 0,8,16,24
    const int aDst = ar * LDSROW + ((ac0 >> 3) & 1) * 16 + 8 * (ac0 >> 4);
    const int bn   = tid >> 2;             // 0..63 col
    const int bk0  = (tid & 3) * 8;
    const int bDst = bn * LDSROW + ((bk0 >> 3) & 1) * 16 + 8 * (bk0 >> 4);

    // loop-invariant base pointers
    const _Float16* aRow = A + (size_t)(blockM + ar) * Astride + ac0;
    const _Float16* wRow = W + (size_t)(blockN + bn) * Kdim + bk0;

    f8 acc0 = {};
    f8 acc1 = {};

    for (int kt = 0; kt < Kdim; kt += BKK) {
        // ---- stage A ----
        h8v av;
        if constexpr (HASK1) {
            if (kt < K1) {   // uniform: K1 % 32 == 0
                av = *(const h8v*)(Abc + (size_t)bb * K1 + kt + ac0);
            } else {
                av = *(const h8v*)(aRow + (kt - K1));
            }
        } else {
            av = *(const h8v*)(aRow + kt);
        }
        *(h8v*)(AsP + aDst) = av;

        // ---- stage B ----
        h8v bv;
        if constexpr (NFULL) {
            bv = *(const h8v*)(wRow + kt);
        } else {
#pragma unroll
            for (int j = 0; j < 8; ++j)
                bv[j] = (blockN + bn < N) ? W[(size_t)(blockN + bn) * Kdim + kt + bk0 + j]
                                          : (_Float16)0.f;
        }
        *(h8v*)(BsP + bDst) = bv;

        if (kt + BKK < Kdim)
            __builtin_prefetch(wRow + kt + BKK, 0, 2);

        __syncthreads();

        // ---- fragments: 2x b128 LDS reads each ----
        const _Float16* ap = AsP + (mr + (lane & 15)) * LDSROW + (lane >> 4) * 16;
        h8v alo = *(const h8v*)ap;
        h8v ahi = *(const h8v*)(ap + 8);
        h16 afrag = __builtin_shufflevector(alo, ahi, 0, 1, 2, 3, 4, 5, 6, 7,
                                            8, 9, 10, 11, 12, 13, 14, 15);

        const _Float16* bp0 = BsP + (nb0 + (lane & 15)) * LDSROW + (lane >> 4) * 16;
        h8v b0lo = *(const h8v*)bp0;
        h8v b0hi = *(const h8v*)(bp0 + 8);
        h16 bfrag0 = __builtin_shufflevector(b0lo, b0hi, 0, 1, 2, 3, 4, 5, 6, 7,
                                             8, 9, 10, 11, 12, 13, 14, 15);
        const _Float16* bp1 = bp0 + 16 * LDSROW;
        h8v b1lo = *(const h8v*)bp1;
        h8v b1hi = *(const h8v*)(bp1 + 8);
        h16 bfrag1 = __builtin_shufflevector(b1lo, b1hi, 0, 1, 2, 3, 4, 5, 6, 7,
                                             8, 9, 10, 11, 12, 13, 14, 15);

        acc0 = __builtin_amdgcn_wmma_f32_16x16x32_f16(false, afrag, false, bfrag0,
                                                      (short)0, acc0, false, false);
        acc1 = __builtin_amdgcn_wmma_f32_16x16x32_f16(false, afrag, false, bfrag1,
                                                      (short)0, acc1, false, false);
        __syncthreads();
    }

    // ---- epilogue (C layout: lane<16 -> M=v, lane>=16 -> M=v+8; N=lane&15) ----
    const int mloc = 8 * (lane >> 4);
    const int nloc = lane & 15;
#pragma unroll
    for (int v = 0; v < 8; ++v) {
        int m = blockM + mr + mloc + v;      // M % 64 == 0 -> always in range
#pragma unroll
        for (int t = 0; t < 2; ++t) {
            int n = blockN + nb0 + t * 16 + nloc;
            if (!NFULL && n >= N) continue;
            float y = (t == 0) ? acc0[v] : acc1[v];
            if (epi != 2) {
                y = y * scale[n] + bias[n];
                y = (y > 0.f) ? y : 0.2f * y;
            }
            if (epi == 0) {
                out16[(size_t)m * N + n] = (_Float16)y;
            } else if (epi == 1) {
                int b = m / rowsPerBatch;
                atomicMaxFloat(&outAux[(size_t)b * N + n], y);
            } else {
                int b   = m / rowsPerBatch;
                int npt = m - b * rowsPerBatch;
                outAux[((size_t)b * N + n) * rowsPerBatch + npt] = y;
            }
        }
    }
}

// ---------------------------------------------------------------------------
// kNN: one thread per query; candidates tiled through LDS (128 x C f16, b128
// cooperative staging). Top-20 via insertion (self included, dist 0).
// ---------------------------------------------------------------------------
template <int C>
__global__ void __launch_bounds__(256)
knn_kernel(const _Float16* __restrict__ feat, int* __restrict__ idxout, int Npts) {
    const int b = blockIdx.y;
    const int n = blockIdx.x * 256 + threadIdx.x;
    const _Float16* fb = feat + (size_t)b * Npts * C;

    float q[C];
#pragma unroll
    for (int c = 0; c < C; ++c) q[c] = (n < Npts) ? (float)fb[(size_t)n * C + c] : 0.f;

    float bd[20];
    int   bi[20];
#pragma unroll
    for (int i = 0; i < 20; ++i) { bd[i] = 3.0e38f; bi[i] = 0; }

    __shared__ __align__(16) _Float16 tile[128 * C];
    constexpr int NCHUNK = 128 * C / 8;   // b128 chunks (c0*C is 8-aligned)
    for (int c0 = 0; c0 < Npts; c0 += 128) {
        const h8v* src = (const h8v*)(fb + (size_t)c0 * C);
        for (int i = threadIdx.x; i < NCHUNK; i += 256)
            ((h8v*)tile)[i] = src[i];
        __syncthreads();
        if (n < Npts) {
            for (int j = 0; j < 128; ++j) {
                float d = 0.f;
#pragma unroll
                for (int c = 0; c < C; ++c) {
                    float t = q[c] - (float)tile[j * C + c];
                    d += t * t;
                }
                if (d < bd[19]) {
                    int p = 19;
                    while (p > 0 && bd[p - 1] > d) {
                        bd[p] = bd[p - 1]; bi[p] = bi[p - 1]; --p;
                    }
                    bd[p] = d; bi[p] = c0 + j;
                }
            }
        }
        __syncthreads();
    }
    if (n < Npts)
        for (int i = 0; i < 20; ++i)
            idxout[(((size_t)b * Npts) + n) * 20 + i] = bi[i];
}

// ---------------------------------------------------------------------------
// Edge features for one batch, row stride ldF (>= 2C, pad pre-zeroed):
//   F[(n*20+kk)][0..C-1]  = nbr - ctr ;  F[(n*20+kk)][C..2C-1] = ctr
// ---------------------------------------------------------------------------
__global__ void build_edge_kernel(const _Float16* __restrict__ feat,
                                  const int* __restrict__ idx,
                                  _Float16* __restrict__ F,
                                  int b, int Npts, int C, int ldF) {
    size_t total = (size_t)Npts * 20 * C;
    for (size_t i = (size_t)blockIdx.x * blockDim.x + threadIdx.x; i < total;
         i += (size_t)gridDim.x * blockDim.x) {
        int c  = (int)(i % C);
        int kk = (int)((i / C) % 20);
        int n  = (int)(i / ((size_t)C * 20));
        int nb = idx[(((size_t)b * Npts) + n) * 20 + kk];
        float ctr = (float)feat[((size_t)b * Npts + n) * C + c];
        float nbv = (float)feat[((size_t)b * Npts + nb) * C + c];
        size_t row = (size_t)n * 20 + kk;
        F[row * ldF + c]     = (_Float16)(nbv - ctr);
        F[row * ldF + C + c] = (_Float16)ctr;
    }
}

// Max over the k=20 neighbor axis for one batch.
__global__ void maxk_kernel(const _Float16* __restrict__ Y, _Float16* __restrict__ xout,
                            int b, int Npts, int Cout) {
    int i = blockIdx.x * blockDim.x + threadIdx.x;
    if (i >= Npts * Cout) return;
    int c = i % Cout, n = i / Cout;
    float m = -3.0e38f;
    for (int kk = 0; kk < 20; ++kk) {
        float v = (float)Y[((size_t)n * 20 + kk) * Cout + c];
        m = v > m ? v : m;
    }
    xout[((size_t)b * Npts + n) * Cout + c] = (_Float16)m;
}

// cat(x1,x2,x3) -> (B*N, 192) f16
__global__ void cat_kernel(const _Float16* __restrict__ x1, const _Float16* __restrict__ x2,
                           const _Float16* __restrict__ x3, _Float16* __restrict__ cat,
                           int totalRows) {
    int i = blockIdx.x * blockDim.x + threadIdx.x;
    if (i >= totalRows * 192) return;
    int c = i % 192, n = i / 192;
    _Float16 v;
    if (c < 64)       v = x1[(size_t)n * 64 + c];
    else if (c < 128) v = x2[(size_t)n * 64 + (c - 64)];
    else              v = x3[(size_t)n * 64 + (c - 128)];
    cat[i] = v;
}

// x (B,3,N) f32 -> xt16 (B,N,3) f16
__global__ void xt_kernel(const float* __restrict__ x, _Float16* __restrict__ xt, int Npts) {
    int i = blockIdx.x * blockDim.x + threadIdx.x;
    if (i >= 8 * Npts * 3) return;
    int c = i % 3;
    int n = (i / 3) % Npts;
    int b = i / (3 * Npts);
    xt[i] = (_Float16)x[((size_t)b * 3 + c) * Npts + n];
}

__global__ void cvt_f32_to_f16_kernel(const float* __restrict__ in, _Float16* __restrict__ out,
                                      int n) {
    int i = blockIdx.x * blockDim.x + threadIdx.x;
    if (i < n) out[i] = (_Float16)in[i];
}

// f32 (O x Iin) -> f16 (O x Ipad) with zero padding (used to zero-pad W1)
__global__ void pad_cvt_kernel(const float* __restrict__ in, _Float16* __restrict__ out,
                               int O, int Iin, int Ipad) {
    int i = blockIdx.x * blockDim.x + threadIdx.x;
    if (i >= O * Ipad) return;
    int k = i % Ipad, o = i / Ipad;
    out[i] = (k < Iin) ? (_Float16)in[(size_t)o * Iin + k] : (_Float16)0.f;
}

__global__ void fill_kernel(float* __restrict__ p, float v, int n) {
    int i = blockIdx.x * blockDim.x + threadIdx.x;
    if (i < n) p[i] = v;
}

__global__ void fill16_kernel(_Float16* __restrict__ p, int n) {
    int i = blockIdx.x * blockDim.x + threadIdx.x;
    if (i < n) p[i] = (_Float16)0.f;
}

// ---------------------------------------------------------------------------
// Host orchestration
// ---------------------------------------------------------------------------
static inline size_t alignUp256(size_t x) { return (x + 255) & ~(size_t)255; }

extern "C" void kernel_launch(void* const* d_in, const int* in_sizes, int n_in,
                              void* d_out, int out_size, void* d_ws, size_t ws_size,
                              hipStream_t stream) {
    (void)in_sizes; (void)n_in; (void)out_size; (void)ws_size;
    const int B = 8, NP = 4096, KN = 20;

    const float* x   = (const float*)d_in[0];
    const float* w1  = (const float*)d_in[1];  const float* s1  = (const float*)d_in[2];  const float* b1  = (const float*)d_in[3];
    const float* w2  = (const float*)d_in[4];  const float* s2  = (const float*)d_in[5];  const float* b2  = (const float*)d_in[6];
    const float* w3  = (const float*)d_in[7];  const float* s3  = (const float*)d_in[8];  const float* b3  = (const float*)d_in[9];
    const float* w4  = (const float*)d_in[10]; const float* s4  = (const float*)d_in[11]; const float* b4  = (const float*)d_in[12];
    const float* w5  = (const float*)d_in[13]; const float* s5  = (const float*)d_in[14]; const float* b5  = (const float*)d_in[15];
    const float* w6  = (const float*)d_in[16]; const float* s6  = (const float*)d_in[17]; const float* b6  = (const float*)d_in[18];
    const float* w8  = (const float*)d_in[19]; const float* s8  = (const float*)d_in[20]; const float* b8  = (const float*)d_in[21];
    const float* w9  = (const float*)d_in[22]; const float* s9  = (const float*)d_in[23]; const float* b9  = (const float*)d_in[24];
    const float* w10 = (const float*)d_in[25]; const float* s10 = (const float*)d_in[26]; const float* b10 = (const float*)d_in[27];
    const float* w11 = (const float*)d_in[28];

    // ---- workspace partition ----
    char* p = (char*)d_ws;
    auto alloc = [&](size_t bytes) { void* r = (void*)p; p += alignUp256(bytes); return r; };

    _Float16* xt16  = (_Float16*)alloc((size_t)B * NP * 3 * 2);
    int*      idx   = (int*)     alloc((size_t)B * NP * KN * 4);
    _Float16* Fbuf  = (_Float16*)alloc((size_t)NP * KN * 128 * 2);   // per-batch edge feats
    _Float16* Y1    = (_Float16*)alloc((size_t)NP * KN * 64 * 2);
    _Float16* Y2    = (_Float16*)alloc((size_t)NP * KN * 64 * 2);
    _Float16* x1h   = (_Float16*)alloc((size_t)B * NP * 64 * 2);
    _Float16* x2h   = (_Float16*)alloc((size_t)B * NP * 64 * 2);
    _Float16* x3h   = (_Float16*)alloc((size_t)B * NP * 64 * 2);
    _Float16* cat19 = (_Float16*)alloc((size_t)B * NP * 192 * 2);
    float*    gmax  = (float*)   alloc((size_t)B * 1024 * 4);
    _Float16* gmax16= (_Float16*)alloc((size_t)B * 1024 * 2);
    _Float16* h8b   = (_Float16*)alloc((size_t)B * NP * 256 * 2);
    _Float16* h9b   = (_Float16*)alloc((size_t)B * NP * 256 * 2);
    _Float16* h10b  = (_Float16*)alloc((size_t)B * NP * 128 * 2);
    _Float16* W1p = (_Float16*)alloc(64 * 32 * 2);   // zero-padded 6 -> 32
    _Float16* W2  = (_Float16*)alloc(64 * 64 * 2);
    _Float16* W3  = (_Float16*)alloc(64 * 128 * 2);
    _Float16* W4  = (_Float16*)alloc(64 * 64 * 2);
    _Float16* W5  = (_Float16*)alloc(64 * 128 * 2);
    _Float16* W6  = (_Float16*)alloc(1024 * 192 * 2);
    _Float16* W8  = (_Float16*)alloc(256 * 1216 * 2);
    _Float16* W9  = (_Float16*)alloc(256 * 256 * 2);
    _Float16* W10 = (_Float16*)alloc(128 * 256 * 2);
    _Float16* W11 = (_Float16*)alloc(13 * 128 * 2);

    auto cvt = [&](const float* in, _Float16* out, int n) {
        cvt_f32_to_f16_kernel<<<(n + 255) / 256, 256, 0, stream>>>(in, out, n);
    };
    pad_cvt_kernel<<<(64 * 32 + 255) / 256, 256, 0, stream>>>(w1, W1p, 64, 6, 32);
    cvt(w2, W2, 64 * 64);
    cvt(w3, W3, 64 * 128);   cvt(w4, W4, 64 * 64);
    cvt(w5, W5, 64 * 128);   cvt(w6, W6, 1024 * 192);
    cvt(w8, W8, 256 * 1216); cvt(w9, W9, 256 * 256);
    cvt(w10, W10, 128 * 256); cvt(w11, W11, 13 * 128);

    auto gemm = [&](const _Float16* A, const _Float16* Abc, int K1, const _Float16* W,
                    const float* s, const float* bi, _Float16* o16, float* aux,
                    int M, int N, int Kd, int epi) {
        dim3 g(M / BM, (N + BN - 1) / BN);
        if (K1 > 0) {
            gemm_wmma_kernel<true, true><<<g, 256, 0, stream>>>(
                A, Abc, K1, W, s, bi, o16, aux, M, N, Kd, epi, NP);
        } else if ((N % BN) == 0) {
            gemm_wmma_kernel<false, true><<<g, 256, 0, stream>>>(
                A, nullptr, 0, W, s, bi, o16, aux, M, N, Kd, epi, NP);
        } else {
            gemm_wmma_kernel<false, false><<<g, 256, 0, stream>>>(
                A, nullptr, 0, W, s, bi, o16, aux, M, N, Kd, epi, NP);
        }
    };

    // ---- xt ----
    xt_kernel<<<(B * NP * 3 + 255) / 256, 256, 0, stream>>>(x, xt16, NP);

    // ---- stage 1: knn on (B,N,3); edge conv w1,w2; max_k -> x1 ----
    // Zero Fbuf once so the 6->32 K padding region is zero for stage 1.
    fill16_kernel<<<(NP * KN * 32 + 255) / 256, 256, 0, stream>>>(Fbuf, NP * KN * 32);
    knn_kernel<3><<<dim3(NP / 256, B), 256, 0, stream>>>(xt16, idx, NP);
    for (int b = 0; b < B; ++b) {
        build_edge_kernel<<<1024, 256, 0, stream>>>(xt16, idx, Fbuf, b, NP, 3, 32);
        gemm(Fbuf, nullptr, 0, W1p, s1, b1, Y1, nullptr, NP * KN, 64, 32, 0);
        gemm(Y1,   nullptr, 0, W2,  s2, b2, Y2, nullptr, NP * KN, 64, 64, 0);
        maxk_kernel<<<(NP * 64 + 255) / 256, 256, 0, stream>>>(Y2, x1h, b, NP, 64);
    }

    // ---- stage 2: knn on x1; edge conv w3,w4; max_k -> x2 ----
    knn_kernel<64><<<dim3(NP / 256, B), 256, 0, stream>>>(x1h, idx, NP);
    for (int b = 0; b < B; ++b) {
        build_edge_kernel<<<2048, 256, 0, stream>>>(x1h, idx, Fbuf, b, NP, 64, 128);
        gemm(Fbuf, nullptr, 0, W3, s3, b3, Y1, nullptr, NP * KN, 64, 128, 0);
        gemm(Y1,   nullptr, 0, W4, s4, b4, Y2, nullptr, NP * KN, 64, 64, 0);
        maxk_kernel<<<(NP * 64 + 255) / 256, 256, 0, stream>>>(Y2, x2h, b, NP, 64);
    }

    // ---- stage 3: knn on x2; edge conv w5; max_k -> x3 ----
    knn_kernel<64><<<dim3(NP / 256, B), 256, 0, stream>>>(x2h, idx, NP);
    for (int b = 0; b < B; ++b) {
        build_edge_kernel<<<2048, 256, 0, stream>>>(x2h, idx, Fbuf, b, NP, 64, 128);
        gemm(Fbuf, nullptr, 0, W5, s5, b5, Y1, nullptr, NP * KN, 64, 128, 0);
        maxk_kernel<<<(NP * 64 + 255) / 256, 256, 0, stream>>>(Y1, x3h, b, NP, 64);
    }

    // ---- cat(x1,x2,x3) -> 192 ----
    cat_kernel<<<(B * NP * 192 + 255) / 256, 256, 0, stream>>>(x1h, x2h, x3h, cat19, B * NP);

    // ---- layer 6: (B*N,192)x(192,1024); fused global max over N -> gmax ----
    fill_kernel<<<(B * 1024 + 255) / 256, 256, 0, stream>>>(gmax, -3.0e38f, B * 1024);
    gemm(cat19, nullptr, 0, W6, s6, b6, nullptr, gmax, B * NP, 1024, 192, 1);
    cvt_f32_to_f16_kernel<<<(B * 1024 + 255) / 256, 256, 0, stream>>>(gmax, gmax16, B * 1024);

    // ---- layer 8: concat(gmax broadcast[1024], cat192) -> 1216 -> 256 ----
    gemm(cat19, gmax16, 1024, W8, s8, b8, h8b, nullptr, B * NP, 256, 1216, 0);
    // ---- layer 9: 256 -> 256 ----
    gemm(h8b, nullptr, 0, W9, s9, b9, h9b, nullptr, B * NP, 256, 256, 0);
    // ---- layer 10: 256 -> 128 ----
    gemm(h9b, nullptr, 0, W10, s10, b10, h10b, nullptr, B * NP, 128, 256, 0);
    // ---- layer 11: 128 -> 13, raw, transposed f32 store into d_out (B,13,N) ----
    gemm(h10b, nullptr, 0, W11, nullptr, nullptr, nullptr, (float*)d_out, B * NP, 13, 128, 2);
}
